// Treecaps_51470888075929
// MI455X (gfx1250) — matrix operand order repaired and checked by reading, not compiled, not run on hardware
//
#include <hip/hip_runtime.h>
#include <hip/hip_bf16.h>
#include <math.h>

typedef _Float16 half_t;
typedef __attribute__((ext_vector_type(8)))  _Float16 v8h;
typedef __attribute__((ext_vector_type(16))) _Float16 v16h;
typedef __attribute__((ext_vector_type(8)))  float    v8f;

#define NN      20000      // nodes
#define CC      32         // max children
#define FF      128        // feature size
#define CFF     128        // conv features
#define NCOV    4
#define KDIM    384        // 3*FF
#define ODIM    512        // NCOV*CFF
#define TOPA    20
#define TOPB    30
#define NCAPS   2560       // TOPA*CFF
#define UROWS   3840       // TOPB*CFF
#define LBL     104
#define OD      8
#define LD      832        // LBL*OD

// ---- workspace layout (bytes) ----
#define OFF_NODES   0u                      // f32 [NN][FF]            10,240,000
#define OFF_H       10240000u               // f16 [NN][KDIM]          15,360,000
#define OFF_WCT     25600000u               // f16 [ODIM][KDIM]           393,216
#define OFF_PRIMT   25993216u               // f32 [ODIM][NN]          40,960,000
#define OFF_U       66953216u               // f32 [TOPB][CFF][NCOV]       61,440
#define OFF_MZ      67014656u               // f32 [2][UROWS]              30,720
#define OFF_S       67045376u               // f32 [NCAPS][NCOV]           40,960
#define OFF_UHAT    67086336u               // f32 [NCAPS][LD]          8,519,680
#define OFF_GAMMA   75606016u               // f32 [NCAPS][LBL]         1,064,960
#define OFF_VJ      76670976u               // f32 [3][LD]                  9,984

// ---------------------------------------------------------------------------
// 0) weight prep: WcT[o=d*128+cf][k=g*128+f] = w_conv[d][g][f][cf]  (f16)
// ---------------------------------------------------------------------------
__global__ __launch_bounds__(256) void k_wprep(const float* __restrict__ w_conv,
                                               half_t* __restrict__ WcT) {
    int g = blockIdx.x * 256 + threadIdx.x;       // < 512*384
    if (g >= ODIM * KDIM) return;
    int o = g / KDIM, k = g - o * KDIM;
    int d = o >> 7, cf = o & 127;
    int gg = k >> 7, f = k & 127;
    WcT[g] = (half_t)w_conv[(((d * 3 + gg) * FF) + f) * CFF + cf];
}

// ---------------------------------------------------------------------------
// 1) embedding gather: nodes[n][f] = Wemd[node_ids[n]][f]
// ---------------------------------------------------------------------------
__global__ __launch_bounds__(256) void k_embed(const int* __restrict__ node_ids,
                                               const float* __restrict__ Wemd,
                                               float* __restrict__ nodes) {
    int g = blockIdx.x * 256 + threadIdx.x;       // < NN*FF
    int n = g >> 7, f = g & 127;
    nodes[g] = Wemd[(size_t)node_ids[n] * FF + f];
}

// ---------------------------------------------------------------------------
// 2) per-node window: H[n] = [self | sum_c cr_c*child_c | sum_c cl_c*child_c]
//    one block (128 threads) per node; coefficients per reference formulas
// ---------------------------------------------------------------------------
__global__ __launch_bounds__(128) void k_window(const int* __restrict__ children,
                                                const float* __restrict__ nodes,
                                                half_t* __restrict__ H) {
    __shared__ int   chs[CC];
    __shared__ float crs[CC], cls[CC];
    __shared__ int   ns_s;
    int n = blockIdx.x, tid = threadIdx.x;

    if (tid < CC) chs[tid] = children[(size_t)n * CC + tid];
    __syncthreads();
    if (tid == 0) {
        int ns = 0;
        for (int c = 0; c < CC; ++c) ns += (chs[c] != 0);
        ns_s = ns;
    }
    __syncthreads();
    if (tid < CC) {
        int c = tid;
        float maskc = (chs[c] != 0) ? 1.0f : 0.0f;
        int ns = ns_s;
        float crv;
        if (ns == 1) crv = (c == 0) ? 0.5f : 0.0f;          // 'singles' path
        else         crv = (float)c * maskc / ((float)ns - 1.0f); // ns==0 -> /-1, mask 0
        crs[c] = crv;
        cls[c] = (1.0f - crv) * maskc;
    }
    __syncthreads();

    float self = nodes[(size_t)n * FF + tid];
    float ar = 0.0f, al = 0.0f;
#pragma unroll 4
    for (int c = 0; c < CC; ++c) {
        int chd = chs[c];
        if (chd != 0) {                                     // lookup row 0 is zero
            float v = nodes[(size_t)chd * FF + tid];
            ar += crs[c] * v;
            al += cls[c] * v;
        }
    }
    size_t base = (size_t)n * KDIM;
    H[base + tid]          = (half_t)self;
    H[base + FF + tid]     = (half_t)ar;
    H[base + 2 * FF + tid] = (half_t)al;
}

// ---------------------------------------------------------------------------
// 3) GEMM: primT[o][m] = tanh( sum_k H[m][k]*WcT[o][k] + b_conv[o] )
//    256 threads = 8 waves, one 16x16 (MxN) tile per wave, K=384 in 12 WMMAs
// ---------------------------------------------------------------------------
__global__ __launch_bounds__(256) void k_gemm(const half_t* __restrict__ H,
                                              const half_t* __restrict__ WcT,
                                              const float* __restrict__ b_conv,
                                              float* __restrict__ primT) {
    int wave = threadIdx.x >> 5;
    int lane = threadIdx.x & 31;
    int t  = blockIdx.x * 8 + wave;     // tile id, 0..39999
    int tN = t & 31;                    // 32 N-tiles
    int tM = t >> 5;                    // 1250 M-tiles
    int hi = lane >> 4;                 // 0/1 lane half
    int m  = tM * 16 + (lane & 15);
    int o  = tN * 16 + (lane & 15);

    const half_t* Ab = H   + (size_t)m * KDIM + hi * 16;  // 16 contiguous K values
    const half_t* Bb = WcT + (size_t)o * KDIM + hi * 16;  // 16 contiguous K values

    v8f acc = {};
#pragma unroll
    for (int kk = 0; kk < 12; ++kk) {
        int kb = kk * 32;
        v8h lo = *(const v8h*)(Ab + kb);
        v8h hh = *(const v8h*)(Ab + kb + 8);
        v16h a;
#pragma unroll
        for (int e = 0; e < 8; ++e) { a[e] = lo[e]; a[8 + e] = hh[e]; }
        v16h b = *(const v16h*)(Bb + kb);
        acc = __builtin_amdgcn_wmma_f32_16x16x32_f16(
            false, a, false, b, (short)0, acc, false, false);
    }

    float bias = b_conv[o];
    float* p = primT + (size_t)o * NN + tM * 16 + hi * 8; // 8 contiguous M per lane
#pragma unroll
    for (int r = 0; r < 8; ++r) p[r] = tanhf(acc[r] + bias);
}

// ---------------------------------------------------------------------------
// 4) top-30 (sorted desc) per channel row; one block (256 thr) per q in [0,512)
//    phase A: per-thread sorted-30 list in LDS; phase B: 30x parallel argmax merge
// ---------------------------------------------------------------------------
__global__ __launch_bounds__(256) void k_topk(const float* __restrict__ primT,
                                              float* __restrict__ U) {
    __shared__ float lists[256 * TOPB];
    __shared__ float cval[256];
    __shared__ int   cth[256];
    int tid = threadIdx.x;
    int q = blockIdx.x;
    const float* row = primT + (size_t)q * NN;
    float* myl = lists + tid * TOPB;

    int cnt = 0;
    float minv = -1e30f;
    for (int idx = tid; idx < NN; idx += 256) {
        float v = row[idx];
        if (cnt < TOPB) {
            int p = cnt;
            while (p > 0 && myl[p - 1] < v) { myl[p] = myl[p - 1]; --p; }
            myl[p] = v; ++cnt;
            minv = myl[cnt - 1];
        } else if (v > minv) {
            int p = TOPB - 1;
            while (p > 0 && myl[p - 1] < v) { myl[p] = myl[p - 1]; --p; }
            myl[p] = v;
            minv = myl[TOPB - 1];
        }
    }
    __syncthreads();

    int head = 0;
    int cf = q & 127, d = q >> 7;
    for (int rank = 0; rank < TOPB; ++rank) {
        cval[tid] = (head < cnt) ? myl[head] : -1e30f;
        cth[tid] = tid;
        __syncthreads();
        for (int s = 128; s > 0; s >>= 1) {
            if (tid < s && cval[tid + s] > cval[tid]) {
                cval[tid] = cval[tid + s]; cth[tid] = cth[tid + s];
            }
            __syncthreads();
        }
        if (tid == 0) U[rank * ODIM + cf * NCOV + d] = cval[0];
        if (tid == cth[0]) ++head;
        __syncthreads();
    }
}

// ---------------------------------------------------------------------------
// 5a) per-u_i-row softmax stats over 2560 alphas: m_i, Z_i
// ---------------------------------------------------------------------------
__global__ __launch_bounds__(256) void k_alpha_stats(const float* __restrict__ U,
                                                     float* __restrict__ mZ) {
    __shared__ float alpha[NCAPS];
    __shared__ float red[256];
    int i = blockIdx.x, tid = threadIdx.x;
    float u0 = U[4 * i], u1 = U[4 * i + 1], u2 = U[4 * i + 2], u3 = U[4 * i + 3];

    float lmax = -1e30f;
    for (int j = tid; j < NCAPS; j += 256) {
        const float* vj = U + 4 * j;
        float a = u0 * vj[0] + u1 * vj[1] + u2 * vj[2] + u3 * vj[3];
        alpha[j] = a;
        lmax = fmaxf(lmax, a);
    }
    red[tid] = lmax; __syncthreads();
    for (int s = 128; s > 0; s >>= 1) {
        if (tid < s) red[tid] = fmaxf(red[tid], red[tid + s]);
        __syncthreads();
    }
    float m = red[0]; __syncthreads();

    float lsum = 0.0f;
    for (int j = tid; j < NCAPS; j += 256) lsum += expf(alpha[j] - m);
    red[tid] = lsum; __syncthreads();
    for (int s = 128; s > 0; s >>= 1) {
        if (tid < s) red[tid] += red[tid + s];
        __syncthreads();
    }
    if (tid == 0) { mZ[i] = m; mZ[UROWS + i] = red[0]; }
}

// ---------------------------------------------------------------------------
// 5b) v_J[j] = sum_i softmax_j(alpha_i)[j] * u_i ; then squash -> static caps
// ---------------------------------------------------------------------------
__global__ __launch_bounds__(256) void k_vts(const float* __restrict__ U,
                                             const float* __restrict__ mZ,
                                             float* __restrict__ s) {
    int j = blockIdx.x * 256 + threadIdx.x;      // < 2560
    float v0 = U[4 * j], v1 = U[4 * j + 1], v2 = U[4 * j + 2], v3 = U[4 * j + 3];
    float a0 = 0, a1 = 0, a2 = 0, a3 = 0;
    for (int i = 0; i < UROWS; ++i) {
        const float* ui = U + 4 * i;
        float al = v0 * ui[0] + v1 * ui[1] + v2 * ui[2] + v3 * ui[3];
        float w = expf(al - mZ[i]) / mZ[UROWS + i];
        a0 += w * ui[0]; a1 += w * ui[1]; a2 += w * ui[2]; a3 += w * ui[3];
    }
    float sq = a0 * a0 + a1 * a1 + a2 * a2 + a3 * a3;
    float scale = sq / (1.0f + sq) / sqrtf(sq + 1e-9f);
    s[4 * j + 0] = a0 * scale; s[4 * j + 1] = a1 * scale;
    s[4 * j + 2] = a2 * scale; s[4 * j + 3] = a3 * scale;
}

// ---------------------------------------------------------------------------
// 6) u_hat[n][ld] = sum_k W_dr[n][ld][k] * s[n][k]
// ---------------------------------------------------------------------------
__global__ __launch_bounds__(256) void k_uhat(const float* __restrict__ W_dr,
                                              const float* __restrict__ s,
                                              float* __restrict__ uhat) {
    int g = blockIdx.x * 256 + threadIdx.x;      // < 2560*832
    int n = g / LD;
    const float4 w = ((const float4*)W_dr)[g];
    const float* sn = s + 4 * n;
    uhat[g] = w.x * sn[0] + w.y * sn[1] + w.z * sn[2] + w.w * sn[3];
}

// ---------------------------------------------------------------------------
// 7a) gamma[n][l] = softmax_l( sum_{p<nprev} dot8(uhat[n][l], vJ[p][l]) )
// ---------------------------------------------------------------------------
__global__ __launch_bounds__(128) void k_gamma(const float* __restrict__ uhat,
                                               const float* __restrict__ vJ,
                                               float* __restrict__ gamma,
                                               int nprev) {
    __shared__ float red[128];
    int n = blockIdx.x, l = threadIdx.x;
    float delta = 0.0f;
    if (l < LBL) {
        const float* uh = uhat + (size_t)n * LD + l * OD;
        for (int p = 0; p < nprev; ++p) {
            const float* v = vJ + p * LD + l * OD;
            float acc = 0.0f;
#pragma unroll
            for (int d = 0; d < OD; ++d) acc += uh[d] * v[d];
            delta += acc;
        }
    }
    red[l] = (l < LBL) ? delta : -1e30f;
    __syncthreads();
    for (int sN = 64; sN > 0; sN >>= 1) {
        if (l < sN) red[l] = fmaxf(red[l], red[l + sN]);
        __syncthreads();
    }
    float m = red[0]; __syncthreads();
    float e = (l < LBL) ? expf(delta - m) : 0.0f;
    red[l] = e; __syncthreads();
    for (int sN = 64; sN > 0; sN >>= 1) {
        if (l < sN) red[l] += red[l + sN];
        __syncthreads();
    }
    float Z = red[0];
    if (l < LBL) gamma[(size_t)n * LBL + l] = e / Z;
}

// ---------------------------------------------------------------------------
// 7b) s_J[l][:] = sum_n gamma[n][l]*uhat[n][l][:] + bias; squash -> vJ[r][l][:]
//     one block per label; last iteration also writes d_out
// ---------------------------------------------------------------------------
__global__ __launch_bounds__(256) void k_sj(const float* __restrict__ uhat,
                                            const float* __restrict__ gamma,
                                            const float* __restrict__ biases,
                                            float* __restrict__ vJ,
                                            float* __restrict__ out,
                                            int r) {
    __shared__ float red[OD * 256];
    int l = blockIdx.x, tid = threadIdx.x;
    float acc[OD];
#pragma unroll
    for (int d = 0; d < OD; ++d) acc[d] = 0.0f;
    for (int n = tid; n < NCAPS; n += 256) {
        float g = gamma[(size_t)n * LBL + l];
        const float* uh = uhat + (size_t)n * LD + l * OD;
#pragma unroll
        for (int d = 0; d < OD; ++d) acc[d] += g * uh[d];
    }
#pragma unroll
    for (int d = 0; d < OD; ++d) red[d * 256 + tid] = acc[d];
    __syncthreads();
    for (int sN = 128; sN > 0; sN >>= 1) {
        if (tid < sN) {
#pragma unroll
            for (int d = 0; d < OD; ++d)
                red[d * 256 + tid] += red[d * 256 + tid + sN];
        }
        __syncthreads();
    }
    if (tid == 0) {
        float sv[OD], sq = 0.0f;
#pragma unroll
        for (int d = 0; d < OD; ++d) {
            sv[d] = red[d * 256] + biases[l * OD + d];
            sq += sv[d] * sv[d];
        }
        float scale = sq / (1.0f + sq) / sqrtf(sq + 1e-9f);
#pragma unroll
        for (int d = 0; d < OD; ++d) {
            float o = sv[d] * scale;
            vJ[r * LD + l * OD + d] = o;
            if (r == 2) out[l * OD + d] = o;
        }
    }
}

// ---------------------------------------------------------------------------
extern "C" void kernel_launch(void* const* d_in, const int* in_sizes, int n_in,
                              void* d_out, int out_size, void* d_ws, size_t ws_size,
                              hipStream_t stream) {
    const int*   node_ids  = (const int*)d_in[0];
    const int*   children  = (const int*)d_in[1];
    const float* Wemd      = (const float*)d_in[2];
    const float* w_conv    = (const float*)d_in[3];
    const float* b_conv    = (const float*)d_in[4];
    const float* W_dr      = (const float*)d_in[5];
    const float* biases_dr = (const float*)d_in[6];
    float* out = (float*)d_out;

    char* ws = (char*)d_ws;
    float*  nodes = (float*)(ws + OFF_NODES);
    half_t* H     = (half_t*)(ws + OFF_H);
    half_t* WcT   = (half_t*)(ws + OFF_WCT);
    float*  primT = (float*)(ws + OFF_PRIMT);
    float*  U     = (float*)(ws + OFF_U);
    float*  mZ    = (float*)(ws + OFF_MZ);
    float*  s     = (float*)(ws + OFF_S);
    float*  uhat  = (float*)(ws + OFF_UHAT);
    float*  gamma = (float*)(ws + OFF_GAMMA);
    float*  vJ    = (float*)(ws + OFF_VJ);

    k_wprep<<<(ODIM * KDIM + 255) / 256, 256, 0, stream>>>(w_conv, WcT);
    k_embed<<<(NN * FF) / 256, 256, 0, stream>>>(node_ids, Wemd, nodes);
    k_window<<<NN, 128, 0, stream>>>(children, nodes, H);
    k_gemm<<<(1250 * 32) / 8, 256, 0, stream>>>(H, WcT, b_conv, primT);
    k_topk<<<ODIM, 256, 0, stream>>>(primT, U);
    k_alpha_stats<<<UROWS, 256, 0, stream>>>(U, mZ);
    k_vts<<<NCAPS / 256, 256, 0, stream>>>(U, mZ, s);
    k_uhat<<<(NCAPS * LD) / 256, 256, 0, stream>>>(W_dr, s, uhat);

    for (int r = 0; r < 3; ++r) {
        k_gamma<<<NCAPS, 128, 0, stream>>>(uhat, vJ, gamma, r);
        k_sj<<<LBL, 256, 0, stream>>>(uhat, gamma, biases_dr, vJ, out, r);
    }
}
